// encoder_45045617001167
// MI455X (gfx1250) — matrix operand order repaired and checked by reading, not compiled, not run on hardware
//
#include <hip/hip_runtime.h>
#include <hip/hip_bf16.h>

// ---------------------------------------------------------------------------
// GAT encoder for MI455X (gfx1250, wave32, WMMA + async/TDM data movement).
//  * all matmuls via v_wmma_f32_16x16x32_bf16, f32 accumulate
//  * all f32->bf16 conversions hoisted out of hot loops (pre-converted /
//    pre-transposed operands), fragments are two aligned ds_load_b128 each
//  * GEMM tiles staged with global_load_async_to_lds_b128 (ASYNCcnt),
//    attention H-tiles staged with the Tensor Data Mover (TENSORcnt),
//    both double-buffered so DMA overlaps WMMA
//  * two-pass softmax: 64MB adjacency is resident in the 192MB L2
// ---------------------------------------------------------------------------

typedef __bf16 bf16_t;
typedef __attribute__((ext_vector_type(16))) __bf16 v16bf;
typedef __attribute__((ext_vector_type(8)))  __bf16 v8bf;
typedef __attribute__((ext_vector_type(8)))  float  v8f;
typedef __attribute__((ext_vector_type(4)))  unsigned int v4u;
typedef __attribute__((ext_vector_type(8)))  int v8i_t;
typedef __attribute__((ext_vector_type(4)))  int v4i_t;

#define GAT_ALPHA 0.2f
#define NEG_FILL  -9.0e15f
#define GN_EPS    1e-5f
#define NN        4096

#if defined(__has_builtin)
#if __has_builtin(__builtin_amdgcn_tensor_load_to_lds)
#define HAVE_TDM 1
#endif
#endif
#ifndef HAVE_TDM
#define HAVE_TDM 0
#endif

// Compile-time probes (show up in stderr; do not affect codegen)
#define TDM_STR2(x) #x
#define TDM_STR(x) TDM_STR2(x)
#pragma message("CDNA5 probe: __clang_major__ = " TDM_STR(__clang_major__))
#if HAVE_TDM
#warning "CDNA5 probe: TDM tensor_load_to_lds path ENABLED in attn_agg_kernel"
#else
#warning "CDNA5 probe: TDM builtin NOT found; async_b128 fallback in attn_agg_kernel"
#endif

// Low 32 bits of a flat LDS pointer == LDS byte offset (ISA 10.2: LDS
// aperture address is truncated to addr[31:0]).
__device__ __forceinline__ unsigned lds_off(const void* p) {
    return (unsigned)(uintptr_t)p;
}

// 16B async global -> LDS copy (per-lane addresses), tracked by ASYNCcnt.
__device__ __forceinline__ void async_b128(const bf16_t* gsrc, bf16_t* ldst) {
    unsigned loff = lds_off(ldst);
    unsigned long long ga = (unsigned long long)(uintptr_t)gsrc;
    asm volatile("global_load_async_to_lds_b128 %0, %1, off"
                 :: "v"(loff), "v"(ga) : "memory");
}
__device__ __forceinline__ void wait_async0() {
#if defined(__has_builtin) && __has_builtin(__builtin_amdgcn_s_wait_asynccnt)
    __builtin_amdgcn_s_wait_asynccnt(0);
#else
    asm volatile("s_wait_asynccnt 0x0" ::: "memory");
#endif
}

#if HAVE_TDM
// TDM: load a 2D tile (32 bf16 cols x `rows` rows, row stride NN elements)
// into LDS with pad_enable giving an 80B LDS row pitch (64B data + 16B pad,
// conflict-free for the b128 fragment reads).
__device__ __forceinline__ void tdm_load_tile(const bf16_t* gsrc, void* ldst, int rows)
{
    unsigned long long ga = (unsigned long long)(uintptr_t)gsrc;
    v4u g0;
    g0[0] = 1u;                                   // count=1 valid descriptor
    g0[1] = lds_off(ldst);                        // lds_addr
    g0[2] = (unsigned)(ga & 0xffffffffu);         // global_addr[31:0]
    g0[3] = (unsigned)((ga >> 32) & 0x1ffffffu)   // global_addr[56:32]
          | (2u << 30);                           // type = 2 ("image")
    v8i_t g1;
    g1[0] = (1 << 16)                             // data_size = 2 bytes
          | (1 << 20)                             // pad_enable
          | (3 << 22)                             // pad_interval: 16 DWORDs
          | (3 << 25);                            // pad_amount: 4 DWORDs
    g1[1] = (int)(4096u << 16);                   // tensor_dim0 = NN (lo16)
    g1[2] = (int)((unsigned)rows << 16);          // tensor_dim1 = rows (lo16)
    g1[3] = (int)(32u << 16);                     // tile_dim0 = 32
    g1[4] = rows;                                 // tile_dim1 = rows
    g1[5] = 4096;                                 // tensor_dim0_stride = NN
    g1[6] = 0; g1[7] = 0;
    v4i_t gz = {0, 0, 0, 0};
#if __clang_major__ >= 23
    v8i_t gz8 = {0, 0, 0, 0, 0, 0, 0, 0};
    __builtin_amdgcn_tensor_load_to_lds(g0, g1, gz, gz, gz8, 0);
#else
    __builtin_amdgcn_tensor_load_to_lds(g0, g1, gz, gz, 0);
#endif
}
__device__ __forceinline__ void wait_tensor0() {
#if __has_builtin(__builtin_amdgcn_s_wait_tensorcnt)
    __builtin_amdgcn_s_wait_tensorcnt(0);
#else
    asm volatile("s_wait_tensorcnt 0x0" ::: "memory");
#endif
}
#endif

__device__ __forceinline__ v16bf frag_cat(v8bf lo, v8bf hi) {
    return __builtin_shufflevector(lo, hi, 0,1,2,3,4,5,6,7,8,9,10,11,12,13,14,15);
}

// ---------------------------------------------------------------------------
// GEMM: C[M,N](f32, opt) / Cb[M,N](bf16, opt) = A[M,K](bf16) @ Bt[N,K](bf16)^T
// (+ bias). Block 256 = 8 waves; tile 128x64; wave tile 16x64.  A/B tiles
// double-buffered in LDS via async b128 copies; 80B pitch, b128 fragments.
// M%128==0, N%64==0, K%32==0.
// ---------------------------------------------------------------------------
__global__ __launch_bounds__(256) void gemm_bf16_kernel(
    const bf16_t* __restrict__ A, const bf16_t* __restrict__ Bt,
    const float* __restrict__ bias, float* __restrict__ C,
    bf16_t* __restrict__ Cb, int M, int N, int K)
{
    __shared__ alignas(16) bf16_t As[2][128 * 40];   // 128 rows x 32K, pitch 40
    __shared__ alignas(16) bf16_t Bs[2][64 * 40];    // 64 cols x 32K

    const int tid  = threadIdx.x;
    const int wave = tid >> 5, lane = tid & 31, lm = lane & 15, hi = lane >> 4;
    const int mblk = blockIdx.x * 128;
    const int n0   = blockIdx.y * 64;

    auto stage = [&](int buf, int k0) {
        #pragma unroll
        for (int it = 0; it < 2; ++it) {             // A: 128 rows x 4 segs
            int id = tid + it * 256;
            int row = id >> 2, sg = id & 3;
            async_b128(A + (size_t)(mblk + row) * K + k0 + sg * 8,
                       &As[buf][row * 40 + sg * 8]);
        }
        {                                            // B: 64 cols x 4 segs
            int col = tid >> 2, sg = tid & 3;
            async_b128(Bt + (size_t)(n0 + col) * K + k0 + sg * 8,
                       &Bs[buf][col * 40 + sg * 8]);
        }
    };

    v8f acc[4];
    #pragma unroll
    for (int c = 0; c < 4; ++c)
        #pragma unroll
        for (int e = 0; e < 8; ++e) acc[c][e] = 0.0f;

    stage(0, 0);
    wait_async0();
    __syncthreads();

    const int rw = wave * 16 + lm;
    for (int k0 = 0, buf = 0; k0 < K; k0 += 32, buf ^= 1) {
        if (k0 + 32 < K) stage(buf ^ 1, k0 + 32);    // DMA next while computing

        v8bf alo = *(const v8bf*)&As[buf][rw * 40 + hi * 8];        // K=8hi..
        v8bf ahi = *(const v8bf*)&As[buf][rw * 40 + 16 + hi * 8];   // K=16+8hi..
        v16bf a = frag_cat(alo, ahi);
        #pragma unroll
        for (int c = 0; c < 4; ++c) {
            int col = c * 16 + lm;
            v8bf blo = *(const v8bf*)&Bs[buf][col * 40 + hi * 8];       // K=8hi..
            v8bf bhi = *(const v8bf*)&Bs[buf][col * 40 + 16 + hi * 8];
            acc[c] = __builtin_amdgcn_wmma_f32_16x16x32_bf16(
                false, a, false, frag_cat(blo, bhi), (short)0, acc[c], false, false);
        }
        wait_async0();
        __syncthreads();
    }

    #pragma unroll
    for (int c = 0; c < 4; ++c) {
        int col = n0 + c * 16 + lm;
        float bv = bias ? bias[col] : 0.0f;
        #pragma unroll
        for (int v = 0; v < 8; ++v) {
            int row = mblk + wave * 16 + v + 8 * hi;
            float val = acc[c][v] + bv;
            if (C)  C [(size_t)row * N + col] = val;
            if (Cb) Cb[(size_t)row * N + col] = (bf16_t)val;
        }
    }
}

// ---------------------------------------------------------------------------
// Tiled transpose + f32->bf16: in[R,C] f32 -> out[C,R] bf16  (R,C % 32 == 0)
// ---------------------------------------------------------------------------
__global__ __launch_bounds__(256) void tcvt_kernel(
    const float* __restrict__ in, bf16_t* __restrict__ out, int R, int C)
{
    __shared__ float t[32][33];
    const int bc = blockIdx.x * 32, br = blockIdx.y * 32;
    const int lx = threadIdx.x & 31, ly = threadIdx.x >> 5;
    for (int yy = ly; yy < 32; yy += 8)
        t[yy][lx] = in[(size_t)(br + yy) * C + bc + lx];
    __syncthreads();
    for (int yy = ly; yy < 32; yy += 8)
        out[(size_t)(bc + yy) * R + br + lx] = (bf16_t)t[lx][yy];
}

__global__ __launch_bounds__(256) void cvt_bf16_kernel(
    const float* __restrict__ in, bf16_t* __restrict__ out, int n)
{
    int idx = blockIdx.x * 256 + threadIdx.x;
    if (idx < n) out[idx] = (bf16_t)in[idx];
}

// ---------------------------------------------------------------------------
// Dual GEMV: s[r] = hW[r,:].a[0:F], d[r] = hW[r,:].a[F:2F]
// ---------------------------------------------------------------------------
__global__ __launch_bounds__(256) void gemv2_kernel(
    const float* __restrict__ H, const float* __restrict__ a,
    float* __restrict__ s, float* __restrict__ d, int F)
{
    __shared__ float rs[256], rd[256];
    const int row = blockIdx.x;
    const float* h = H + (size_t)row * F;
    float ps = 0.0f, pd = 0.0f;
    for (int j = threadIdx.x; j < F; j += 256) {
        float hv = h[j];
        ps += hv * a[j];
        pd += hv * a[F + j];
    }
    rs[threadIdx.x] = ps; rd[threadIdx.x] = pd;
    __syncthreads();
    for (int off = 128; off > 0; off >>= 1) {
        if (threadIdx.x < off) {
            rs[threadIdx.x] += rs[threadIdx.x + off];
            rd[threadIdx.x] += rd[threadIdx.x + off];
        }
        __syncthreads();
    }
    if (threadIdx.x == 0) { s[row] = rs[0]; d[row] = rd[0]; }
}

// ---------------------------------------------------------------------------
// Softmax pass 1: m[i] = max_j efill, l[i] = sum_j exp(efill - m)
// ---------------------------------------------------------------------------
__global__ __launch_bounds__(256) void softmax_stats_kernel(
    const float* __restrict__ s, const float* __restrict__ d,
    const int* __restrict__ adj, float* __restrict__ m, float* __restrict__ l)
{
    __shared__ float red[256];
    const int i = blockIdx.x;
    const float si = s[i];
    const int* arow = adj + (size_t)i * NN;

    float ef[16];
    float mx = -3.0e38f;
    #pragma unroll
    for (int t = 0; t < 16; ++t) {
        int j = threadIdx.x + t * 256;
        float e = si + d[j];
        e = e > 0.0f ? e : GAT_ALPHA * e;
        e = (arow[j] > 0) ? e : NEG_FILL;
        ef[t] = e;
        mx = fmaxf(mx, e);
    }
    red[threadIdx.x] = mx; __syncthreads();
    for (int off = 128; off > 0; off >>= 1) {
        if (threadIdx.x < off)
            red[threadIdx.x] = fmaxf(red[threadIdx.x], red[threadIdx.x + off]);
        __syncthreads();
    }
    mx = red[0]; __syncthreads();

    float sum = 0.0f;
    #pragma unroll
    for (int t = 0; t < 16; ++t) sum += __expf(ef[t] - mx);
    red[threadIdx.x] = sum; __syncthreads();
    for (int off = 128; off > 0; off >>= 1) {
        if (threadIdx.x < off) red[threadIdx.x] += red[threadIdx.x + off];
        __syncthreads();
    }
    if (threadIdx.x == 0) { m[i] = mx; l[i] = red[0]; }
}

// ---------------------------------------------------------------------------
// Softmax pass 2: out[i,f] = act((1/l_i) * sum_j exp(e_ij - m_i) * hW[j,f])
// Ht is hW transposed, bf16 [F, NN].  Block = 16-row tile, 8 waves each own
// an F/8 column slice.  H tile (32 x F) staged by TDM (or async fallback),
// P tile built in LDS; both double-buffered.  F is a template constant so
// the WMMA fragment loop fully unrolls.
// ---------------------------------------------------------------------------
template <int F>
__global__ __launch_bounds__(256) void attn_agg_kernel(
    const float* __restrict__ s, const float* __restrict__ dvec,
    const float* __restrict__ mrow, const float* __restrict__ lrow,
    const int* __restrict__ adj, const bf16_t* __restrict__ Ht,
    float* __restrict__ out, bf16_t* __restrict__ outb,
    int ldout, int coloff, int act)
{
    __shared__ alignas(16) bf16_t Hs[2][F * 40];   // F rows x 32K, pitch 40
    __shared__ alignas(16) bf16_t Ps[2][16 * 40];  // 16 rows x 32K
    __shared__ float sI[16], mI[16], ilI[16];

    const int tid  = threadIdx.x;
    const int wave = tid >> 5, lane = tid & 31, lm = lane & 15, hi = lane >> 4;
    const int i0   = blockIdx.x * 16;
    constexpr int FW = F / 8;        // per-wave F slice (64 or 32)
    constexpr int NFRAG = FW / 16;   // 4 or 2
    const int f0 = wave * FW;

    auto stageH = [&](int buf, int j0) {
#if HAVE_TDM
        if (wave == 0) tdm_load_tile(Ht + j0, &Hs[buf][0], F);
#else
        for (int id = tid; id < F * 4; id += 256) {
            int f = id >> 2, sg = id & 3;
            async_b128(Ht + (size_t)f * NN + j0 + sg * 8, &Hs[buf][f * 40 + sg * 8]);
        }
#endif
    };
    auto waitH = [&]() {
#if HAVE_TDM
        if (wave == 0) wait_tensor0();
#else
        wait_async0();
#endif
    };
    auto buildP = [&](int buf, int j0) {
        #pragma unroll
        for (int t = 0; t < 2; ++t) {
            int idx = tid + t * 256;     // 512 = 16 x 32
            int r = idx >> 5, c = idx & 31;
            int j = j0 + c;
            float e = sI[r] + dvec[j];
            e = e > 0.0f ? e : GAT_ALPHA * e;
            e = (adj[(size_t)(i0 + r) * NN + j] > 0) ? e : NEG_FILL;
            Ps[buf][r * 40 + c] = (bf16_t)__expf(e - mI[r]);
        }
    };

    if (tid < 16) {
        sI[tid]  = s[i0 + tid];
        mI[tid]  = mrow[i0 + tid];
        ilI[tid] = 1.0f / lrow[i0 + tid];
    }
    __syncthreads();

    v8f acc[NFRAG];
    #pragma unroll
    for (int c = 0; c < NFRAG; ++c)
        #pragma unroll
        for (int e = 0; e < 8; ++e) acc[c][e] = 0.0f;

    stageH(0, 0);
    buildP(0, 0);
    waitH();
    __syncthreads();

    for (int j0 = 0, buf = 0; j0 < NN; j0 += 32, buf ^= 1) {
        if (j0 + 32 < NN) {              // DMA + P build for next tile overlap
            stageH(buf ^ 1, j0 + 32);    // the WMMAs on the current tile
            buildP(buf ^ 1, j0 + 32);
        }
        v8bf plo = *(const v8bf*)&Ps[buf][lm * 40 + hi * 8];
        v8bf phi = *(const v8bf*)&Ps[buf][lm * 40 + 16 + hi * 8];
        v16bf a = frag_cat(plo, phi);
        #pragma unroll
        for (int c = 0; c < NFRAG; ++c) {
            int fr = f0 + c * 16 + lm;
            v8bf blo = *(const v8bf*)&Hs[buf][fr * 40 + hi * 8];
            v8bf bhi = *(const v8bf*)&Hs[buf][fr * 40 + 16 + hi * 8];
            acc[c] = __builtin_amdgcn_wmma_f32_16x16x32_bf16(
                false, a, false, frag_cat(blo, bhi), (short)0, acc[c], false, false);
        }
        waitH();
        __syncthreads();
    }

    #pragma unroll
    for (int c = 0; c < NFRAG; ++c) {
        int col = coloff + f0 + c * 16 + lm;
        #pragma unroll
        for (int v = 0; v < 8; ++v) {
            int r = v + 8 * hi;
            float val = acc[c][v] * ilI[r];
            if (act == 1) val = fmaxf(val, 0.0f);
            if (out)  out [(size_t)(i0 + r) * ldout + col] = val;
            if (outb) outb[(size_t)(i0 + r) * ldout + col] = (bf16_t)val;
        }
    }
}

// ---------------------------------------------------------------------------
// GroupNorm(1 group) over all elements, unbiased variance; deterministic.
// ---------------------------------------------------------------------------
__global__ __launch_bounds__(256) void gn_rowsum_kernel(
    const float* __restrict__ h, float* __restrict__ part, int C)
{
    __shared__ float rs[256], rq[256];
    const int row = blockIdx.x;
    const float* p = h + (size_t)row * C;
    float sv = 0.0f, qv = 0.0f;
    for (int c = threadIdx.x; c < C; c += 256) { float v = p[c]; sv += v; qv += v * v; }
    rs[threadIdx.x] = sv; rq[threadIdx.x] = qv; __syncthreads();
    for (int off = 128; off > 0; off >>= 1) {
        if (threadIdx.x < off) {
            rs[threadIdx.x] += rs[threadIdx.x + off];
            rq[threadIdx.x] += rq[threadIdx.x + off];
        }
        __syncthreads();
    }
    if (threadIdx.x == 0) { part[row * 2] = rs[0]; part[row * 2 + 1] = rq[0]; }
}

__global__ __launch_bounds__(256) void gn_finalize_kernel(
    const float* __restrict__ part, float* __restrict__ stats, int rows, float cnt)
{
    __shared__ float rs[256], rq[256];
    float sv = 0.0f, qv = 0.0f;
    for (int r = threadIdx.x; r < rows; r += 256) { sv += part[r * 2]; qv += part[r * 2 + 1]; }
    rs[threadIdx.x] = sv; rq[threadIdx.x] = qv; __syncthreads();
    for (int off = 128; off > 0; off >>= 1) {
        if (threadIdx.x < off) {
            rs[threadIdx.x] += rs[threadIdx.x + off];
            rq[threadIdx.x] += rq[threadIdx.x + off];
        }
        __syncthreads();
    }
    if (threadIdx.x == 0) {
        float mu  = rs[0] / cnt;
        float var = (rq[0] - rs[0] * rs[0] / cnt) / (cnt - 1.0f);  // ddof=1
        stats[0] = mu;
        stats[1] = rsqrtf(var + GN_EPS);
    }
}

__global__ __launch_bounds__(256) void gn_apply_kernel(
    float* __restrict__ h, bf16_t* __restrict__ hb, const float* __restrict__ stats,
    const float* __restrict__ w, const float* __restrict__ b, int C, int total)
{
    int idx = blockIdx.x * 256 + threadIdx.x;
    if (idx >= total) return;
    int c = idx % C;
    float v = (h[idx] - stats[0]) * stats[1] * w[c] + b[c];
    h[idx]  = v;
    hb[idx] = (bf16_t)v;
}

// ---------------------------------------------------------------------------
// Host orchestration
// ---------------------------------------------------------------------------
extern "C" void kernel_launch(void* const* d_in, const int* in_sizes, int n_in,
                              void* d_out, int out_size, void* d_ws, size_t ws_size,
                              hipStream_t stream)
{
    (void)in_sizes; (void)n_in; (void)out_size; (void)ws_size;
    const float* x   = (const float*)d_in[0];
    const int*   adj = (const int*)  d_in[1];
    const float* l1w = (const float*)d_in[2];
    const float* l1b = (const float*)d_in[3];
    const float* l2w = (const float*)d_in[4];
    const float* l2b = (const float*)d_in[5];
    const float* c1W = (const float*)d_in[6];
    const float* c1a = (const float*)d_in[7];
    const float* aW  = (const float*)d_in[8];
    const float* aa  = (const float*)d_in[9];
    const float* gnw = (const float*)d_in[10];
    const float* gnb = (const float*)d_in[11];
    const float* c2W = (const float*)d_in[12];
    const float* c2a = (const float*)d_in[13];
    float* out = (float*)d_out;

    const int N = NN;
    char* ws = (char*)d_ws;
    size_t off = 0;
    auto alloc = [&](size_t bytes) -> void* {
        void* p = ws + off;
        off = (off + bytes + 255) & ~(size_t)255;
        return p;
    };

    float*  hw   = (float*) alloc((size_t)N * 512 * 4);    // hW (f32) per GAT
    bf16_t* Htb  = (bf16_t*)alloc((size_t)512 * N * 2);    // hW^T bf16 [F,N]
    bf16_t* xb   = (bf16_t*)alloc((size_t)N * 128 * 2);
    bf16_t* hAb  = (bf16_t*)alloc((size_t)N * 64 * 2);
    bf16_t* hBb  = (bf16_t*)alloc((size_t)N * 64 * 2);
    bf16_t* g1b  = (bf16_t*)alloc((size_t)N * 512 * 2);    // conv1 out (bf16)
    float*  cat  = (float*) alloc((size_t)N * 1536 * 4);   // head concat / GN
    bf16_t* catb = (bf16_t*)alloc((size_t)N * 1536 * 2);
    bf16_t* l1wt = (bf16_t*)alloc((size_t)64 * 128 * 2);   // weights^T bf16
    bf16_t* l2wt = (bf16_t*)alloc((size_t)64 * 64 * 2);
    bf16_t* c1Wt = (bf16_t*)alloc((size_t)512 * 64 * 2);
    bf16_t* aWt  = (bf16_t*)alloc((size_t)6 * 256 * 512 * 2);
    bf16_t* c2Wt = (bf16_t*)alloc((size_t)256 * 1536 * 2);
    float*  sv   = (float*) alloc((size_t)N * 4);
    float*  dv   = (float*) alloc((size_t)N * 4);
    float*  mv   = (float*) alloc((size_t)N * 4);
    float*  lv   = (float*) alloc((size_t)N * 4);
    float*  part = (float*) alloc((size_t)N * 2 * 4);
    float*  stat = (float*) alloc(2 * 4);

    // one-time operand preparation (bf16 conversions / transposes)
    cvt_bf16_kernel<<<(N * 128 + 255) / 256, 256, 0, stream>>>(x, xb, N * 128);
    tcvt_kernel<<<dim3(64 / 32, 128 / 32), 256, 0, stream>>>(l1w, l1wt, 128, 64);
    tcvt_kernel<<<dim3(64 / 32, 64 / 32),  256, 0, stream>>>(l2w, l2wt, 64, 64);
    tcvt_kernel<<<dim3(512 / 32, 64 / 32), 256, 0, stream>>>(c1W, c1Wt, 64, 512);
    for (int h = 0; h < 6; ++h)
        tcvt_kernel<<<dim3(256 / 32, 512 / 32), 256, 0, stream>>>(
            aW + (size_t)h * 512 * 256, aWt + (size_t)h * 256 * 512, 512, 256);
    tcvt_kernel<<<dim3(256 / 32, 1536 / 32), 256, 0, stream>>>(c2W, c2Wt, 1536, 256);

    auto run_gat = [&](const bf16_t* Ab, int Kin, const bf16_t* Wt, const float* avec,
                       int F, float* outF, bf16_t* outB, int ldo, int co, int act) {
        gemm_bf16_kernel<<<dim3(N / 128, F / 64), 256, 0, stream>>>(
            Ab, Wt, nullptr, hw, nullptr, N, F, Kin);
        tcvt_kernel<<<dim3(F / 32, N / 32), 256, 0, stream>>>(hw, Htb, N, F);
        gemv2_kernel<<<N, 256, 0, stream>>>(hw, avec, sv, dv, F);
        softmax_stats_kernel<<<N, 256, 0, stream>>>(sv, dv, adj, mv, lv);
        if (F == 512)
            attn_agg_kernel<512><<<N / 16, 256, 0, stream>>>(
                sv, dv, mv, lv, adj, Htb, outF, outB, ldo, co, act);
        else
            attn_agg_kernel<256><<<N / 16, 256, 0, stream>>>(
                sv, dv, mv, lv, adj, Htb, outF, outB, ldo, co, act);
    };

    // lin1, lin2 (affine only) -> bf16 activations directly
    gemm_bf16_kernel<<<dim3(N / 128, 1), 256, 0, stream>>>(
        xb, l1wt, l1b, nullptr, hAb, N, 64, 128);
    gemm_bf16_kernel<<<dim3(N / 128, 1), 256, 0, stream>>>(
        hAb, l2wt, l2b, nullptr, hBb, N, 64, 64);

    // conv1: GAT(64->512), relu; only bf16 copy needed downstream
    run_gat(hBb, 64, c1Wt, c1a, 512, nullptr, g1b, 512, 0, 1);

    // 6 heads: GAT(512->256), relu(elu(x)) == relu(x); concat into cat (f32)
    for (int h = 0; h < 6; ++h)
        run_gat(g1b, 512, aWt + (size_t)h * 256 * 512, aa + (size_t)h * 512,
                256, cat, nullptr, 1536, h * 256, 1);

    // GroupNorm over all N*1536 elements; dual f32/bf16 output
    gn_rowsum_kernel<<<N, 256, 0, stream>>>(cat, part, 1536);
    gn_finalize_kernel<<<1, 256, 0, stream>>>(part, stat, N, (float)N * 1536.0f);
    gn_apply_kernel<<<(N * 1536 + 255) / 256, 256, 0, stream>>>(
        cat, catb, stat, gnw, gnb, 1536, N * 1536);

    // conv2: GAT(1536->256), no activation, straight to d_out
    run_gat(catb, 1536, c2Wt, c2a, 256, out, nullptr, 256, 0, 0);
}
